// MultiHeadAttention_37881611551116
// MI455X (gfx1250) — compile-verified, matrix operand order checked by reading, not run on hardware
//
#include <hip/hip_runtime.h>
#include <hip/hip_bf16.h>
#include <stdint.h>

typedef __attribute__((ext_vector_type(16))) __bf16 v16bf;
typedef __attribute__((ext_vector_type(8)))  float  v8f;
typedef uint32_t u32;
typedef unsigned short u16;

#define D_MODEL 1024
#define S_LEN   2048
#define BATCH   4
#define NHEAD   16
#define DK      64
#define M_TOTAL (BATCH * S_LEN)   /* 8192 */

// ---------- helpers ----------

__device__ __forceinline__ u16 f2bf(float f) {
  u32 u = __float_as_uint(f);
  u32 r = u + 0x7FFFu + ((u >> 16) & 1u);   // round-to-nearest-even
  return (u16)(r >> 16);
}

// Low 32 bits of a flat pointer into LDS == wave-relative LDS byte offset
// (ISA: aperture lives entirely in ADDR[63:32]; the LDS offset is the low 32 bits).
__device__ __forceinline__ u32 lds_off(const void* p) {
  return (u32)(uintptr_t)p;
}

// CDNA5 async global->LDS copy, GVS mode (SGPR base + 32-bit VGPR offset).
// Tracked by ASYNCcnt.
__device__ __forceinline__ void async_copy_b128(u32 loff, const void* gbase, u32 goff) {
  asm volatile("global_load_async_to_lds_b128 %0, %1, %2 offset:0"
               :: "v"(loff), "v"(goff), "s"(gbase) : "memory");
}
__device__ __forceinline__ void wait_async0() {
  asm volatile("s_wait_asynccnt 0" ::: "memory");
}
// Leave up to 2 async copies (the prefetch for the next tile) in flight;
// ASYNCcnt retires in order, so this guarantees the current tile is resident.
__device__ __forceinline__ void wait_async2() {
  asm volatile("s_wait_asynccnt 2" ::: "memory");
}

union Frag { uint4 q[2]; v16bf v; };

// Load a 16x32 bf16 WMMA A/B fragment for this lane.
// Per ISA layout: lane holds row (lane%16); VGPRs 0..3 = K chunk (hi?8:0)+0..7,
// VGPRs 4..7 = K chunk 16+(hi?8:0)+0..7  -> two contiguous 16B loads.
__device__ __forceinline__ v16bf frag_ld(const u16* p, int hi) {
  Frag f;
  f.q[0] = *(const uint4*)(p + hi * 8);
  f.q[1] = *(const uint4*)(p + 16 + hi * 8);
  return f.v;
}

// ---------- fp32 -> bf16 cast ----------

__global__ __launch_bounds__(256)
void cast_f32_bf16(const float* __restrict__ in, u16* __restrict__ out, int n) {
  int i = (blockIdx.x * blockDim.x + threadIdx.x) * 4;
  if (i + 3 < n) {
    float4 f = *(const float4*)(in + i);
    ushort4 v;
    v.x = f2bf(f.x); v.y = f2bf(f.y); v.z = f2bf(f.z); v.w = f2bf(f.w);
    *(ushort4*)(out + i) = v;
  } else {
    for (; i < n; ++i) out[i] = f2bf(in[i]);
  }
}

// ---------- bf16 GEMM: C[M,1024] = A[M,1024] x W[1024,1024]^T ----------
// Block = 128x128 tile, BK = 64, double-buffered LDS via async copies.
// 8 waves: 2 (M) x 4 (N); each wave owns 4x2 WMMA 16x16 tiles.

template <bool OUT_F32>
__global__ __launch_bounds__(256)
void gemm_bf16(const u16* __restrict__ A, const u16* __restrict__ W,
               u16* __restrict__ Cb, float* __restrict__ Cf) {
  __shared__ u16 As[2][128 * 64];
  __shared__ u16 Bs[2][128 * 64];

  const int tid  = threadIdx.x;
  const int lane = tid & 31;
  const int hi   = (lane >> 4) & 1;
  const int ln   = lane & 15;
  const int wave = tid >> 5;
  const int wm   = (wave & 1) * 64;
  const int wn   = (wave >> 1) * 32;
  const int m0   = blockIdx.y * 128;
  const int n0   = blockIdx.x * 128;

  auto load_tiles = [&](int buf, int k0) {
#pragma unroll
    for (int i = 0; i < 4; ++i) {               // A: 128 rows x 8 chunks
      int c = tid + 256 * i;
      int row = c >> 3, kc = c & 7;
      u32 goff = (u32)((((m0 + row) * D_MODEL + k0) * 2) + kc * 16);
      async_copy_b128(lds_off(&As[buf][row * 64]) + kc * 16, A, goff);
    }
#pragma unroll
    for (int i = 0; i < 4; ++i) {               // B: rows of W (n index)
      int c = tid + 256 * i;
      int row = c >> 3, kc = c & 7;
      u32 goff = (u32)((((n0 + row) * D_MODEL + k0) * 2) + kc * 16);
      async_copy_b128(lds_off(&Bs[buf][row * 64]) + kc * 16, W, goff);
    }
  };

  v8f acc[4][2] = {};

  load_tiles(0, 0);
  wait_async0();
  __syncthreads();

  const int NT = D_MODEL / 64;                  // 16 K-iterations
  for (int kk = 0; kk < NT; ++kk) {
    int cur = kk & 1;
    if (kk + 1 < NT) load_tiles(cur ^ 1, (kk + 1) * 64);

    // Batch all fragment loads, then run all 16 WMMAs back-to-back.
    v16bf af[2][4], bfm[2][2];
#pragma unroll
    for (int ks = 0; ks < 2; ++ks) {
#pragma unroll
      for (int mi = 0; mi < 4; ++mi)
        af[ks][mi] = frag_ld(&As[cur][(wm + mi * 16 + ln) * 64 + ks * 32], hi);
#pragma unroll
      for (int ni = 0; ni < 2; ++ni)
        bfm[ks][ni] = frag_ld(&Bs[cur][(wn + ni * 16 + ln) * 64 + ks * 32], hi);
    }
#pragma unroll
    for (int ks = 0; ks < 2; ++ks)
#pragma unroll
      for (int mi = 0; mi < 4; ++mi)
#pragma unroll
        for (int ni = 0; ni < 2; ++ni)
          acc[mi][ni] = __builtin_amdgcn_wmma_f32_16x16x32_bf16(
              false, af[ks][mi], false, bfm[ks][ni], (short)0, acc[mi][ni], false, false);

    wait_async0();
    __syncthreads();
  }

#pragma unroll
  for (int mi = 0; mi < 4; ++mi)
#pragma unroll
    for (int ni = 0; ni < 2; ++ni)
#pragma unroll
      for (int r = 0; r < 8; ++r) {
        int m = m0 + wm + mi * 16 + r + 8 * hi;
        int n = n0 + wn + ni * 16 + ln;
        if (OUT_F32) Cf[m * D_MODEL + n] = acc[mi][ni][r];
        else         Cb[m * D_MODEL + n] = f2bf(acc[mi][ni][r]);
      }
}

// ---------- flash attention ----------
// grid (S/128, H, B); 8 waves; each wave owns 16 q-rows. Key blocks of 64.
// Double-buffered K (async copies) and V (register prefetch + transposed LDS
// store) so the next block's loads overlap the current block's WMMAs.

__global__ __launch_bounds__(256)
void attn_kernel(const u16* __restrict__ Qb, const u16* __restrict__ Kb,
                 const u16* __restrict__ Vb, u16* __restrict__ Ob) {
  __shared__ u16 Ks[2][64 * 64];     // [kpos][d]
  __shared__ u16 Vs[2][64 * 64];     // transposed: [d][kpos]
  __shared__ u16 Ps[8][16 * 64];     // wave-private P tiles

  const int tid  = threadIdx.x;
  const int lane = tid & 31;
  const int hi   = (lane >> 4) & 1;
  const int ln   = lane & 15;
  const int wave = tid >> 5;
  const int b    = blockIdx.z;
  const int h    = blockIdx.y;
  const int qbase = blockIdx.x * 128 + wave * 16;

  // Q fragments held in registers for the whole kernel (16x64, two k-steps)
  const int rowQ = (b * S_LEN + qbase + ln) * D_MODEL + h * DK;
  v16bf qf[2];
#pragma unroll
  for (int ks = 0; ks < 2; ++ks)
    qf[ks] = frag_ld(&Qb[rowQ + ks * 32], hi);

  float mrow[8], lrow[8];
#pragma unroll
  for (int r = 0; r < 8; ++r) { mrow[r] = -1e30f; lrow[r] = 0.f; }
  v8f o[4] = {};

  const float scale = 0.125f;        // 1/sqrt(d_k)

  // per-thread copy coordinates (constant across blocks)
  const int c0row = tid >> 3,        c0kc = tid & 7;
  const int c1row = (tid + 256) >> 3, c1kc = (tid + 256) & 7;

  auto issueK = [&](int buf, int kbase) {
    u32 g0 = (u32)((((b * S_LEN + kbase + c0row) * D_MODEL + h * DK) * 2) + c0kc * 16);
    async_copy_b128(lds_off(&Ks[buf][c0row * 64]) + c0kc * 16, Kb, g0);
    u32 g1 = (u32)((((b * S_LEN + kbase + c1row) * D_MODEL + h * DK) * 2) + c1kc * 16);
    async_copy_b128(lds_off(&Ks[buf][c1row * 64]) + c1kc * 16, Kb, g1);
  };
  auto loadV = [&](int kbase, uint4* vr) {
    vr[0] = *(const uint4*)&Vb[(b * S_LEN + kbase + c0row) * D_MODEL + h * DK + c0kc * 8];
    vr[1] = *(const uint4*)&Vb[(b * S_LEN + kbase + c1row) * D_MODEL + h * DK + c1kc * 8];
  };
  auto storeV = [&](int buf, const uint4* vr) {
    union { uint4 q; u16 e[8]; } uu;
    uu.q = vr[0];
#pragma unroll
    for (int j = 0; j < 8; ++j) Vs[buf][(c0kc * 8 + j) * 64 + c0row] = uu.e[j];
    uu.q = vr[1];
#pragma unroll
    for (int j = 0; j < 8; ++j) Vs[buf][(c1kc * 8 + j) * 64 + c1row] = uu.e[j];
  };

  // prologue: start block 0's copies
  issueK(0, 0);
  uint4 vreg[2];
  loadV(0, vreg);

  for (int kb = 0; kb < S_LEN / 64; ++kb) {
    const int cur = kb & 1;

    storeV(cur, vreg);                       // V tile (transposed) into LDS
    const bool more = (kb + 1 < S_LEN / 64);
    if (more) {
      issueK(cur ^ 1, (kb + 1) * 64);        // prefetch next K tile (async)
      loadV((kb + 1) * 64, vreg);            // prefetch next V tile (regs)
    }
    if (more) wait_async2(); else wait_async0();  // current K tile resident
    __syncthreads();

    // scores: S = Q x K^T  (16x64 per wave) — batch loads, then 8 WMMAs
    v16bf kf[2][4];
#pragma unroll
    for (int ks = 0; ks < 2; ++ks)
#pragma unroll
      for (int ni = 0; ni < 4; ++ni)
        kf[ks][ni] = frag_ld(&Ks[cur][(ni * 16 + ln) * 64 + ks * 32], hi);
    v8f sc[4] = {};
#pragma unroll
    for (int ks = 0; ks < 2; ++ks)
#pragma unroll
      for (int ni = 0; ni < 4; ++ni)
        sc[ni] = __builtin_amdgcn_wmma_f32_16x16x32_bf16(
            false, qf[ks], false, kf[ks][ni], (short)0, sc[ni], false, false);

    // online softmax per row (half-wave owns rows r / r+8)
#pragma unroll
    for (int r = 0; r < 8; ++r) {
      float t = -1e30f;
#pragma unroll
      for (int ni = 0; ni < 4; ++ni) { sc[ni][r] *= scale; t = fmaxf(t, sc[ni][r]); }
#pragma unroll
      for (int off = 8; off > 0; off >>= 1) t = fmaxf(t, __shfl_xor(t, off, 16));
      float mnew = fmaxf(mrow[r], t);
      float corr = __expf(mrow[r] - mnew);
      mrow[r] = mnew;
      float rs = 0.f;
#pragma unroll
      for (int ni = 0; ni < 4; ++ni) {
        float p = __expf(sc[ni][r] - mnew);
        sc[ni][r] = p;
        rs += p;
      }
#pragma unroll
      for (int off = 8; off > 0; off >>= 1) rs += __shfl_xor(rs, off, 16);
      lrow[r] = lrow[r] * corr + rs;
#pragma unroll
      for (int ni = 0; ni < 4; ++ni) o[ni][r] *= corr;
    }

    // spill P (bf16) to wave-private LDS: C-layout -> row-major 16x64
#pragma unroll
    for (int ni = 0; ni < 4; ++ni)
#pragma unroll
      for (int r = 0; r < 8; ++r)
        Ps[wave][(r + 8 * hi) * 64 + ni * 16 + ln] = f2bf(sc[ni][r]);

    // O += P x V  (K=64 keys) — batch loads, then 8 WMMAs
    v16bf pf[2], vf[2][4];
#pragma unroll
    for (int ks = 0; ks < 2; ++ks) {
      pf[ks] = frag_ld(&Ps[wave][ln * 64 + ks * 32], hi);
#pragma unroll
      for (int ni = 0; ni < 4; ++ni)
        vf[ks][ni] = frag_ld(&Vs[cur][(ni * 16 + ln) * 64 + ks * 32], hi);
    }
#pragma unroll
    for (int ks = 0; ks < 2; ++ks)
#pragma unroll
      for (int ni = 0; ni < 4; ++ni)
        o[ni] = __builtin_amdgcn_wmma_f32_16x16x32_bf16(
            false, pf[ks], false, vf[ks][ni], (short)0, o[ni], false, false);

    __syncthreads();
  }

  // normalize + store bf16 (B,S,H*dk) concat layout
#pragma unroll
  for (int r = 0; r < 8; ++r) {
    float inv = 1.f / lrow[r];
    int m = qbase + r + 8 * hi;
#pragma unroll
    for (int ni = 0; ni < 4; ++ni)
      Ob[(b * S_LEN + m) * D_MODEL + h * DK + ni * 16 + ln] = f2bf(o[ni][r] * inv);
  }
}

// ---------- launch ----------

extern "C" void kernel_launch(void* const* d_in, const int* in_sizes, int n_in,
                              void* d_out, int out_size, void* d_ws, size_t ws_size,
                              hipStream_t stream) {
  const float* q  = (const float*)d_in[0];
  const float* k  = (const float*)d_in[1];
  const float* v  = (const float*)d_in[2];
  const float* wq = (const float*)d_in[3];
  const float* wk = (const float*)d_in[4];
  const float* wv = (const float*)d_in[5];
  const float* wo = (const float*)d_in[6];
  float* out = (float*)d_out;

  const size_t X  = (size_t)M_TOTAL * D_MODEL;     // 8388608 elems
  const size_t Wn = (size_t)D_MODEL * D_MODEL;     // 1048576 elems

  size_t off = 0;
  char* base = (char*)d_ws;
  auto alloc = [&](size_t bytes) -> u16* {
    u16* p = (u16*)(base + off);
    off += (bytes + 255) & ~(size_t)255;
    return p;
  };
  u16* qb  = alloc(X * 2);
  u16* kb  = alloc(X * 2);
  u16* vb  = alloc(X * 2);
  u16* wqb = alloc(Wn * 2);
  u16* wkb = alloc(Wn * 2);
  u16* wvb = alloc(Wn * 2);
  u16* wob = alloc(Wn * 2);
  u16* Qp  = alloc(X * 2);
  u16* Kp  = alloc(X * 2);
  u16* Vp  = alloc(X * 2);
  u16* Ab  = qb;   // reuse: qb is dead after the Q projection

  dim3 blk(256);
  // casts
  cast_f32_bf16<<<dim3((u32)(X / 4 / 256)), blk, 0, stream>>>(q, qb, (int)X);
  cast_f32_bf16<<<dim3((u32)(X / 4 / 256)), blk, 0, stream>>>(k, kb, (int)X);
  cast_f32_bf16<<<dim3((u32)(X / 4 / 256)), blk, 0, stream>>>(v, vb, (int)X);
  cast_f32_bf16<<<dim3((u32)(Wn / 4 / 256)), blk, 0, stream>>>(wq, wqb, (int)Wn);
  cast_f32_bf16<<<dim3((u32)(Wn / 4 / 256)), blk, 0, stream>>>(wk, wkb, (int)Wn);
  cast_f32_bf16<<<dim3((u32)(Wn / 4 / 256)), blk, 0, stream>>>(wv, wvb, (int)Wn);
  cast_f32_bf16<<<dim3((u32)(Wn / 4 / 256)), blk, 0, stream>>>(wo, wob, (int)Wn);

  // projections
  dim3 ggrid(D_MODEL / 128, M_TOTAL / 128);          // (8, 64)
  gemm_bf16<false><<<ggrid, blk, 0, stream>>>(qb, wqb, Qp, nullptr);
  gemm_bf16<false><<<ggrid, blk, 0, stream>>>(kb, wkb, Kp, nullptr);
  gemm_bf16<false><<<ggrid, blk, 0, stream>>>(vb, wvb, Vp, nullptr);

  // attention
  dim3 agrid(S_LEN / 128, NHEAD, BATCH);             // (16, 16, 4)
  attn_kernel<<<agrid, blk, 0, stream>>>(Qp, Kp, Vp, Ab);

  // output projection (fp32 out)
  gemm_bf16<true><<<ggrid, blk, 0, stream>>>(Ab, wob, nullptr, out);
}
